// LSTMAttentionDot_42322607735638
// MI455X (gfx1250) — compile-verified, matrix operand order checked by reading, not run on hardware
//
#include <hip/hip_runtime.h>
#include <hip/hip_bf16.h>
#include <cstdint>

#define BDIM 64
#define TDIM 64
#define SDIM 512
#define IDIM 512
#define HDIM 512
#define G4H  2048   /* 4*H */
#define STILE 32    /* attention S-tile rows staged in LDS */

typedef __attribute__((ext_vector_type(16))) __bf16 v16bf;
typedef __attribute__((ext_vector_type(8)))  __bf16 v8bf;
typedef __attribute__((ext_vector_type(8)))  float  v8f;

// A/B fragment loader for v_wmma_f32_16x16x32_bf16, row-major along K.
// ISA 7.12.2 (16-bit A 16x32): lane = m%16 | half<<4 holds K runs
// [8*half, 8*half+8) in VGPRs 0-3 and [16+8*half, +8) in VGPRs 4-7,
// i.e. two contiguous 16-byte loads. B uses the mirrored layout with n.
__device__ __forceinline__ v16bf load_frag(const __bf16* rowbase_k0, int half) {
  const v8bf lo = *reinterpret_cast<const v8bf*>(rowbase_k0 + 8 * half);
  const v8bf hi = *reinterpret_cast<const v8bf*>(rowbase_k0 + 16 + 8 * half);
  return __builtin_shufflevector(lo, hi, 0,1,2,3,4,5,6,7,8,9,10,11,12,13,14,15);
}

__device__ __forceinline__ v8f wmma_bf16(v16bf a, v16bf b, v8f c) {
  return __builtin_amdgcn_wmma_f32_16x16x32_bf16(false, a, false, b, (short)0, c,
                                                 false, false);
}

__device__ __forceinline__ float sigm(float x) { return 1.0f / (1.0f + __expf(-x)); }

// ---------------- one-time conversion / init kernels ----------------

__global__ void f32_to_bf16_kernel(const float* __restrict__ src,
                                   __bf16* __restrict__ dst, int n) {
  for (int i = blockIdx.x * blockDim.x + threadIdx.x; i < n;
       i += gridDim.x * blockDim.x)
    dst[i] = (__bf16)src[i];
}

// input (B,T,I) f32 -> (T,B,I) bf16
__global__ void input_tb_bf16_kernel(const float* __restrict__ src,
                                     __bf16* __restrict__ dst) {
  const int n = BDIM * TDIM * IDIM;
  for (int i = blockIdx.x * blockDim.x + threadIdx.x; i < n;
       i += gridDim.x * blockDim.x) {
    int k = i % IDIM;
    int r = i / IDIM;          // r = t*B + b
    int b = r % BDIM;
    int t = r / BDIM;
    dst[i] = (__bf16)src[((size_t)b * TDIM + t) * IDIM + k];
  }
}

// ctx (S,B,H) f32 -> (B,S,H) bf16
__global__ void ctx_tr_bf16_kernel(const float* __restrict__ src,
                                   __bf16* __restrict__ dst) {
  const int n = SDIM * BDIM * HDIM;
  for (int i = blockIdx.x * blockDim.x + threadIdx.x; i < n;
       i += gridDim.x * blockDim.x) {
    int h = i % HDIM;
    int r = i / HDIM;          // r = b*S + s
    int s = r % SDIM;
    int b = r / SDIM;
    dst[i] = (__bf16)src[((size_t)s * BDIM + b) * HDIM + h];
  }
}

__global__ void init_kernel(const float* __restrict__ h0, const float* __restrict__ c0,
                            const float* __restrict__ b_ih, const float* __restrict__ b_hh,
                            __bf16* __restrict__ h_bf0, float* __restrict__ c_f0,
                            float* __restrict__ bsum) {
  int i = blockIdx.x * blockDim.x + threadIdx.x;
  if (i < BDIM * HDIM) {
    h_bf0[i] = (__bf16)h0[i];
    c_f0[i] = c0[i];
  }
  if (i < G4H) bsum[i] = b_ih[i] + b_hh[i];
}

// ---------------- precompute: Xp = x @ W_ih^T + (b_ih+b_hh) ----------------
// M = T*B = 4096, N = 4H = 2048, K = I = 512. grid = (16, 64), 256 thr.
__global__ __launch_bounds__(256) void xp_gemm_kernel(
    const __bf16* __restrict__ xbf, const __bf16* __restrict__ wih,
    const float* __restrict__ bsum, float* __restrict__ Xp) {
  const int lane = threadIdx.x & 31, wave = threadIdx.x >> 5;
  const int half = lane >> 4, l16 = lane & 15;
  const int ncol = blockIdx.x * 128 + wave * 16 + l16;
  const int mbase = blockIdx.y * 64;
  const float bias = bsum[ncol];
  v8f acc[4];
#pragma unroll
  for (int mt = 0; mt < 4; ++mt)
#pragma unroll
    for (int v = 0; v < 8; ++v) acc[mt][v] = bias;
  for (int k0 = 0; k0 < IDIM; k0 += 32) {
    v16bf bfrag = load_frag(wih + (size_t)ncol * IDIM + k0, half);
#pragma unroll
    for (int mt = 0; mt < 4; ++mt) {
      v16bf afrag = load_frag(xbf + (size_t)(mbase + mt * 16 + l16) * IDIM + k0, half);
      acc[mt] = wmma_bf16(afrag, bfrag, acc[mt]);
    }
  }
#pragma unroll
  for (int mt = 0; mt < 4; ++mt)
#pragma unroll
    for (int v = 0; v < 8; ++v) {
      int r = mbase + mt * 16 + v + 8 * half;
      Xp[(size_t)r * G4H + ncol] = acc[mt][v];
    }
}

// ---------------- per-step: gates = Xp[t] + h @ W_hh^T ; LSTM cell ----------
// grid = 16 (column chunks of 32 per gate section), 256 thr.
__global__ __launch_bounds__(256) void lstm_step_kernel(
    const __bf16* __restrict__ h_in, const float* __restrict__ c_in,
    const float* __restrict__ Xp_t, const __bf16* __restrict__ whh,
    float* __restrict__ c_out, float* __restrict__ ht_f,
    __bf16* __restrict__ h_out) {
  __shared__ float ldsG[64 * 128];
  const int lane = threadIdx.x & 31, wave = threadIdx.x >> 5;
  const int half = lane >> 4, l16 = lane & 15;
  const int c0 = blockIdx.x * 32;
  const int sec = wave >> 1, nts = wave & 1;
  const int colg = sec * 512 + c0 + nts * 16 + l16;  // column in (B,4H) gates
  v8f acc[4];
#pragma unroll
  for (int mt = 0; mt < 4; ++mt)
#pragma unroll
    for (int v = 0; v < 8; ++v)
      acc[mt][v] = Xp_t[(size_t)(mt * 16 + v + 8 * half) * G4H + colg];
  for (int k0 = 0; k0 < HDIM; k0 += 32) {
    v16bf bfrag = load_frag(whh + (size_t)colg * HDIM + k0, half);
#pragma unroll
    for (int mt = 0; mt < 4; ++mt) {
      v16bf afrag = load_frag(h_in + (size_t)(mt * 16 + l16) * HDIM + k0, half);
      acc[mt] = wmma_bf16(afrag, bfrag, acc[mt]);
    }
  }
  const int lcol = sec * 32 + nts * 16 + l16;
#pragma unroll
  for (int mt = 0; mt < 4; ++mt)
#pragma unroll
    for (int v = 0; v < 8; ++v)
      ldsG[(mt * 16 + v + 8 * half) * 128 + lcol] = acc[mt][v];
  __syncthreads();
  // fused LSTM cell on the 64x32 hidden-column chunk
  for (int p = threadIdx.x; p < 64 * 32; p += 256) {
    int b = p >> 5, j = p & 31;
    float iv = ldsG[b * 128 + j];
    float fv = ldsG[b * 128 + 32 + j];
    float gv = ldsG[b * 128 + 64 + j];
    float ov = ldsG[b * 128 + 96 + j];
    int hidx = b * HDIM + c0 + j;
    float cx = c_in[hidx];
    float ct = sigm(fv) * cx + sigm(iv) * tanhf(gv);
    float ht = sigm(ov) * tanhf(ct);
    c_out[hidx] = ct;
    ht_f[hidx] = ht;
    h_out[hidx] = (__bf16)ht;
  }
}

// ---------------- per-step fused attention (one workgroup per batch) -------
// Single pass over ctx with online softmax: each 32-row S-tile is read from
// L2 once (during the score pass it is staged into LDS) and the weighted-
// context accumulation replays it from LDS. Halves the dominant L2 traffic.
__global__ __launch_bounds__(256) void attn_fused_kernel(
    const float* __restrict__ ht, const uint32_t* __restrict__ ctx_u32,
    __bf16* __restrict__ wc_bf) {
  __shared__ float    sh_h[512];
  __shared__ uint32_t sh_tile[STILE * 256];  // 32 rows x 512 bf16 = 32 KB
  __shared__ float    sh_sc[STILE];
  __shared__ float    sh_w[STILE];
  const int b = blockIdx.x;
  const int tid = threadIdx.x;
  const int lane = tid & 31, wave = tid >> 5;
  sh_h[tid] = ht[b * 512 + tid];
  sh_h[tid + 256] = ht[b * 512 + tid + 256];
  __syncthreads();

  float m = -1e30f;          // running max
  float l = 0.0f;            // running sum of exp
  float a0 = 0.0f, a1 = 0.0f;  // running weighted context, cols tid / tid+256
  const int w0 = tid >> 1, w1 = (tid + 256) >> 1;
  const bool hi_half = (tid & 1);

  for (int s0 = 0; s0 < SDIM; s0 += STILE) {
    // scores for tile; stage ctx rows into LDS while reading them
#pragma unroll
    for (int j = 0; j < STILE / 8; ++j) {
      int sl = wave * (STILE / 8) + j;
      const uint32_t* row = ctx_u32 + (size_t)(b * 512 + s0 + sl) * 256;
      float partial = 0.f;
#pragma unroll
      for (int i = 0; i < 8; ++i) {
        uint32_t u = row[lane + 32 * i];
        sh_tile[sl * 256 + lane + 32 * i] = u;
        float f0 = __builtin_bit_cast(float, u << 16);
        float f1 = __builtin_bit_cast(float, u & 0xffff0000u);
        int h2 = 2 * (lane + 32 * i);
        partial += f0 * sh_h[h2] + f1 * sh_h[h2 + 1];
      }
#pragma unroll
      for (int off = 16; off >= 1; off >>= 1) partial += __shfl_xor(partial, off, 32);
      if (lane == 0) sh_sc[sl] = partial;
    }
    __syncthreads();
    // tile max (redundant per-thread, deterministic)
    float tm = sh_sc[0];
#pragma unroll
    for (int s = 1; s < STILE; ++s) tm = fmaxf(tm, sh_sc[s]);
    float new_m = fmaxf(m, tm);
    float scale = __expf(m - new_m);
    if (tid < STILE) sh_w[tid] = __expf(sh_sc[tid] - new_m);
    __syncthreads();
    a0 *= scale;
    a1 *= scale;
    float tsum = 0.f;
#pragma unroll 4
    for (int s = 0; s < STILE; ++s) {
      float wgt = sh_w[s];
      tsum += wgt;
      uint32_t u0 = sh_tile[s * 256 + w0];
      uint32_t u1 = sh_tile[s * 256 + w1];
      float c0v = __builtin_bit_cast(float, hi_half ? (u0 & 0xffff0000u) : (u0 << 16));
      float c1v = __builtin_bit_cast(float, hi_half ? (u1 & 0xffff0000u) : (u1 << 16));
      a0 += wgt * c0v;
      a1 += wgt * c1v;
    }
    l = l * scale + tsum;
    m = new_m;
    __syncthreads();   // protect sh_sc/sh_tile before next tile
  }
  const float inv = 1.0f / l;
  a0 *= inv;
  a1 *= inv;
  wc_bf[b * 512 + tid] = (__bf16)a0;
  wc_bf[b * 512 + tid + 256] = (__bf16)a1;
}

// ---------------- per-step: h_tilde = tanh(cat(wc,ht) @ W_out^T) -----------
// M = 64, N = 512, K = 1024. grid = 8, 256 thr. Concat via A-pointer select.
__global__ __launch_bounds__(256) void outproj_kernel(
    const __bf16* __restrict__ wc_bf, const __bf16* __restrict__ ht_bf,
    const __bf16* __restrict__ wout, float* __restrict__ out, int t) {
  const int lane = threadIdx.x & 31, wave = threadIdx.x >> 5;
  const int half = lane >> 4, l16 = lane & 15;
  const int nt = wave >> 1;
  const int mtb = (wave & 1) * 2;
  const int ncol = blockIdx.x * 64 + nt * 16 + l16;
  v8f acc[2];
#pragma unroll
  for (int i = 0; i < 2; ++i)
#pragma unroll
    for (int v = 0; v < 8; ++v) acc[i][v] = 0.0f;
  for (int k0 = 0; k0 < 1024; k0 += 32) {
    v16bf bfrag = load_frag(wout + (size_t)ncol * 1024 + k0, half);
    const __bf16* abase = (k0 < 512) ? (wc_bf + k0) : (ht_bf + (k0 - 512));
#pragma unroll
    for (int i = 0; i < 2; ++i) {
      int mt = mtb + i;
      v16bf afrag = load_frag(abase + (size_t)(mt * 16 + l16) * 512, half);
      acc[i] = wmma_bf16(afrag, bfrag, acc[i]);
    }
  }
#pragma unroll
  for (int i = 0; i < 2; ++i) {
    int mt = mtb + i;
#pragma unroll
    for (int v = 0; v < 8; ++v) {
      int bb = mt * 16 + v + 8 * half;
      out[((size_t)bb * TDIM + t) * HDIM + ncol] = tanhf(acc[i][v]);
    }
  }
}

// ---------------- host ----------------

extern "C" void kernel_launch(void* const* d_in, const int* in_sizes, int n_in,
                              void* d_out, int out_size, void* d_ws, size_t ws_size,
                              hipStream_t stream) {
  const float* input = (const float*)d_in[0];
  const float* h0    = (const float*)d_in[1];
  const float* c0    = (const float*)d_in[2];
  const float* ctx   = (const float*)d_in[3];
  const float* W_ih  = (const float*)d_in[4];
  const float* b_ih  = (const float*)d_in[5];
  const float* W_hh  = (const float*)d_in[6];
  const float* b_hh  = (const float*)d_in[7];
  const float* W_out = (const float*)d_in[8];
  float* out = (float*)d_out;
  (void)in_sizes; (void)n_in; (void)out_size; (void)ws_size;

  uint8_t* ws = (uint8_t*)d_ws;
  size_t off = 0;
  auto carve = [&](size_t bytes) -> void* {
    void* p = ws + off;
    off = (off + bytes + 255) & ~(size_t)255;
    return p;
  };
  float*  Xp   = (float*) carve((size_t)TDIM * BDIM * G4H * 4);
  __bf16* xbf  = (__bf16*)carve((size_t)TDIM * BDIM * IDIM * 2);
  __bf16* ctxT = (__bf16*)carve((size_t)BDIM * SDIM * HDIM * 2);
  __bf16* wih  = (__bf16*)carve((size_t)G4H * IDIM * 2);
  __bf16* whh  = (__bf16*)carve((size_t)G4H * HDIM * 2);
  __bf16* wout = (__bf16*)carve((size_t)HDIM * 2 * HDIM * 2);
  float*  bsum = (float*) carve((size_t)G4H * 4);
  __bf16* h_bf = (__bf16*)carve((size_t)2 * BDIM * HDIM * 2);
  float*  c_f  = (float*) carve((size_t)2 * BDIM * HDIM * 4);
  float*  ht_f = (float*) carve((size_t)BDIM * HDIM * 4);
  __bf16* wcbf = (__bf16*)carve((size_t)BDIM * HDIM * 2);

  f32_to_bf16_kernel<<<512, 256, 0, stream>>>(W_ih, wih, G4H * IDIM);
  f32_to_bf16_kernel<<<512, 256, 0, stream>>>(W_hh, whh, G4H * HDIM);
  f32_to_bf16_kernel<<<512, 256, 0, stream>>>(W_out, wout, HDIM * 2 * HDIM);
  input_tb_bf16_kernel<<<1024, 256, 0, stream>>>(input, xbf);
  ctx_tr_bf16_kernel<<<4096, 256, 0, stream>>>(ctx, ctxT);
  init_kernel<<<(BDIM * HDIM + 255) / 256, 256, 0, stream>>>(h0, c0, b_ih, b_hh,
                                                             h_bf, c_f, bsum);
  xp_gemm_kernel<<<dim3(16, 64), 256, 0, stream>>>(xbf, wih, bsum, Xp);

  const int BH = BDIM * HDIM;
  for (int t = 0; t < TDIM; ++t) {
    const __bf16* hin  = h_bf + (t & 1) * BH;
    __bf16*       hout = h_bf + ((t + 1) & 1) * BH;
    const float*  cin  = c_f + (t & 1) * BH;
    float*        cout = c_f + ((t + 1) & 1) * BH;
    lstm_step_kernel<<<16, 256, 0, stream>>>(hin, cin, Xp + (size_t)t * BDIM * G4H,
                                             whh, cout, ht_f, hout);
    attn_fused_kernel<<<BDIM, 256, 0, stream>>>(ht_f, (const uint32_t*)ctxT, wcbf);
    outproj_kernel<<<8, 256, 0, stream>>>(wcbf, hout, wout, out, t);
  }
  // hT = last ht; cT ended in parity 0 after 64 steps
  hipMemcpyAsync(out + (size_t)BDIM * TDIM * HDIM, ht_f, BH * sizeof(float),
                 hipMemcpyDeviceToDevice, stream);
  hipMemcpyAsync(out + (size_t)BDIM * TDIM * HDIM + BH, c_f, BH * sizeof(float),
                 hipMemcpyDeviceToDevice, stream);
}
